// LocalNonLocalBlock_36326833389655
// MI455X (gfx1250) — matrix operand order repaired and checked by reading, not compiled, not run on hardware
//
#include <hip/hip_runtime.h>
#include <hip/hip_bf16.h>
#include <math.h>

// =====================================================================
// LocalNonLocalBlock (tri-directional Mamba + GDFN) for MI455X / gfx1250
//
// Projections -> v_wmma_f32_16x16x32_f16. One wave computes a 16x32 C
// tile: two WMMA accumulators share one A fragment (2 wmma / K-step).
// Interior tiles run a software-pipelined main loop (next K-tile global
// loads issued under the current WMMAs); edges/K-tail use a branch-free
// clamp-address + multiply-mask guarded step (loads stay unconditional).
// Selective scan parallelized over (b,d,n) chains with shfl reduction
// over the 16 SSM states; dt projection fused in.
// Workspace requirement: ~123 MB (overlapped regions, see layout below).
// =====================================================================

typedef __attribute__((ext_vector_type(16))) _Float16 v16h;
typedef __attribute__((ext_vector_type(8)))  _Float16 v8h;
typedef __attribute__((ext_vector_type(8)))  float    v8f;

#define NB 2
#define NC 64
#define NH 128
#define NW 128
#define NSTATE 16

// ---- branch-free guarded staging: clamp address, multiply-mask value --
// The mask multiply keeps every loaded value unconditionally used, so
// clang cannot sink the loads into exec-masked regions.
static __device__ __forceinline__ void stage_guarded(
    const float* __restrict__ A, const float* __restrict__ Wt,
    int M, int N, int K, int m0, int n0, int k0, int lane,
    _Float16* bufA, _Float16* bufB)
{
    const int ra = lane >> 1, ca = (lane & 1) * 16;
    const int gm = m0 + ra, gn = n0 + lane;
    const int gmc = (gm < M) ? gm : (M - 1);
    const int gnc = (gn < N) ? gn : (N - 1);
    const float mA = (gm < M) ? 1.f : 0.f;         // row masks (cndmask)
    const float mB = (gn < N) ? 1.f : 0.f;
    const float* pa = A  + (size_t)gmc * K;
    const float* pb = Wt + (size_t)gnc * K;
#pragma unroll
    for (int c = 0; c < 16; ++c) {
        const int gk  = k0 + ca + c;
        const int gkc = (gk < K) ? gk : (K - 1);
        const float mk = (gk < K) ? mA : 0.f;
        bufA[c] = (_Float16)(pa[gkc] * mk);        // load always in-bounds
    }
#pragma unroll
    for (int c = 0; c < 32; ++c) {
        const int gk  = k0 + c;
        const int gkc = (gk < K) ? gk : (K - 1);
        const float mk = (gk < K) ? mB : 0.f;
        bufB[c] = (_Float16)(pb[gkc] * mk);        // load always in-bounds
    }
}

// ---- LDS stage + fragment build + 2x WMMA for one K-step -------------
static __device__ __forceinline__ void wmma_step(
    _Float16* sA, _Float16* sB, const _Float16* bufA, const _Float16* bufB,
    int lane, v8f& acc0, v8f& acc1)
{
    const int lh = lane >> 4, lrow = lane & 15;
    const int ra = lane >> 1, ca = (lane & 1) * 16;
    *(v8h*)&sA[ra * 32 + ca]     = *(const v8h*)&bufA[0];
    *(v8h*)&sA[ra * 32 + ca + 8] = *(const v8h*)&bufA[8];
    *(v8h*)&sB[lane * 32 +  0]   = *(const v8h*)&bufB[0];
    *(v8h*)&sB[lane * 32 +  8]   = *(const v8h*)&bufB[8];
    *(v8h*)&sB[lane * 32 + 16]   = *(const v8h*)&bufB[16];
    *(v8h*)&sB[lane * 32 + 24]   = *(const v8h*)&bufB[24];
    __syncthreads();
    v16h fa, fb0, fb1;
#pragma unroll
    for (int e = 0; e < 16; ++e) {
        const int vv = e >> 1, ss = e & 1;
        const int kA = lh * 16 + vv * 2 + ss;
        fa[e]  = sA[lrow * 32 + kA];
        fb0[e] = sB[lrow * 32 + lh * 16 + e];
        fb1[e] = sB[(16 + lrow) * 32 + lh * 16 + e];
    }
    acc0 = __builtin_amdgcn_wmma_f32_16x16x32_f16(
               false, fa, false, fb0, (short)0, acc0, false, false);
    acc1 = __builtin_amdgcn_wmma_f32_16x16x32_f16(
               false, fa, false, fb1, (short)0, acc1, false, false);
    __syncthreads();
}

// ---------------------------------------------------------------------
// GEMM: C[M,N] = A[M,K] * Wt[N,K]^T (+ bias[n]); f32 in/out, f16 WMMA.
// One wave (32 threads) computes a 16(M) x 32(N) tile of C.
// ---------------------------------------------------------------------
__global__ void k_gemm_wmma(const float* __restrict__ A,
                            const float* __restrict__ Wt,
                            const float* __restrict__ bias,
                            float* __restrict__ C,
                            int M, int N, int K)
{
    __shared__ _Float16 sA[16 * 32];   // [m][k]
    __shared__ _Float16 sB[32 * 32];   // [n][k] (two 16-wide n-tiles)

    const int lane = threadIdx.x;          // 0..31
    const int lh   = lane >> 4;            // lane half
    const int lrow = lane & 15;
    const int m0   = blockIdx.x * 16;
    const int n0   = blockIdx.y * 32;
    const bool mn_full = (m0 + 16 <= M) && (n0 + 32 <= N);

    v8f acc0 = {}, acc1 = {};
    const int kfull = K & ~31;

    if (mn_full) {
        // -------- interior tile: pipelined main loop over full K steps ----
        const int ra = lane >> 1, ca = (lane & 1) * 16;
        const float* pa = A  + (size_t)(m0 + ra) * K + ca;
        const float* pb = Wt + (size_t)(n0 + lane) * K;

        _Float16 bufA[16], bufB[32];
        if (kfull > 0) {
#pragma unroll
            for (int c = 0; c < 16; ++c) bufA[c] = (_Float16)pa[c];
#pragma unroll
            for (int c = 0; c < 32; ++c) bufB[c] = (_Float16)pb[c];
        }
        for (int k0 = 0; k0 < kfull; k0 += 32) {
            // stage current tile (128-bit ds stores)
            *(v8h*)&sA[ra * 32 + ca]     = *(v8h*)&bufA[0];
            *(v8h*)&sA[ra * 32 + ca + 8] = *(v8h*)&bufA[8];
            *(v8h*)&sB[lane * 32 +  0]   = *(v8h*)&bufB[0];
            *(v8h*)&sB[lane * 32 +  8]   = *(v8h*)&bufB[8];
            *(v8h*)&sB[lane * 32 + 16]   = *(v8h*)&bufB[16];
            *(v8h*)&sB[lane * 32 + 24]   = *(v8h*)&bufB[24];
            __syncthreads();

            // prefetch next K tile; loads fly under the WMMAs below
            const bool more = (k0 + 32 < kfull);
            _Float16 nxtA[16], nxtB[32];
            if (more) {
                const float* qa = pa + k0 + 32;
                const float* qb = pb + k0 + 32;
#pragma unroll
                for (int c = 0; c < 16; ++c) nxtA[c] = (_Float16)qa[c];
#pragma unroll
                for (int c = 0; c < 32; ++c) nxtB[c] = (_Float16)qb[c];
            }

            // fragments (contiguous k order k = 16*half + e, shared by A and B):
            // A 16x32: lane holds row M=lane&15; lanes 0-15 K=0..15,
            //          lanes 16-31 K=16..31 (2 per VGPR).
            // B 32x16: lane holds col N=lane&15; lanes 0-15 K=0..15, 16-31 K=16..31.
            v16h fa, fb0, fb1;
#pragma unroll
            for (int e = 0; e < 16; ++e) {
                const int vv = e >> 1, ss = e & 1;
                const int kA = lh * 16 + vv * 2 + ss;
                fa[e]  = sA[lrow * 32 + kA];
                fb0[e] = sB[lrow * 32 + lh * 16 + e];
                fb1[e] = sB[(16 + lrow) * 32 + lh * 16 + e];
            }
            acc0 = __builtin_amdgcn_wmma_f32_16x16x32_f16(
                       false, fa, false, fb0, (short)0, acc0, false, false);
            acc1 = __builtin_amdgcn_wmma_f32_16x16x32_f16(
                       false, fa, false, fb1, (short)0, acc1, false, false);
            __syncthreads();

            if (more) {
#pragma unroll
                for (int c = 0; c < 16; ++c) bufA[c] = nxtA[c];
#pragma unroll
                for (int c = 0; c < 32; ++c) bufB[c] = nxtB[c];
            }
        }
        // K tail (e.g. K=170)
        if (kfull < K) {
            _Float16 tA[16], tB[32];
            stage_guarded(A, Wt, M, N, K, m0, n0, kfull, lane, tA, tB);
            wmma_step(sA, sB, tA, tB, lane, acc0, acc1);
        }
    } else {
        // -------- edge tile: branch-free guarded path --------
        for (int k0 = 0; k0 < K; k0 += 32) {
            _Float16 tA[16], tB[32];
            stage_guarded(A, Wt, M, N, K, m0, n0, k0, lane, tA, tB);
            wmma_step(sA, sB, tA, tB, lane, acc0, acc1);
        }
    }

    // writeback: VGPR r -> M = r (lanes 0-15) / 8+r (lanes 16-31), N = lane&15
    const int gn0 = n0 + lrow;
    const int gn1 = gn0 + 16;
    const float bv0 = (bias != nullptr && gn0 < N) ? bias[gn0] : 0.f;
    const float bv1 = (bias != nullptr && gn1 < N) ? bias[gn1] : 0.f;
#pragma unroll
    for (int rr = 0; rr < 8; ++rr) {
        const int gm = m0 + lh * 8 + rr;
        if (gm < M && gn0 < N) C[(size_t)gm * N + gn0] = acc0[rr] + bv0;
        if (gm < M && gn1 < N) C[(size_t)gm * N + gn1] = acc1[rr] + bv1;
    }
}

// ---------------------------------------------------------------------
// Gather one of the three sequence layouts out of x(B,C,H,W) and LayerNorm
// over the feature dim. One block per (b,l) row; blockDim = d (64/128).
// ---------------------------------------------------------------------
__global__ void k_gather_ln(const float* __restrict__ x,
                            const float* __restrict__ lw,
                            const float* __restrict__ lb,
                            float* __restrict__ out,
                            int L, int d, int mode)
{
    __shared__ float s1[128];
    __shared__ float s2[128];
    const int row = blockIdx.x;
    const int b   = row / L;
    const int l   = row % L;
    const int j   = threadIdx.x;
    const size_t CHW = (size_t)NC * NH * NW;

    size_t idx;
    if (mode == 0)      idx = b * CHW + (size_t)j * (NH * NW) + l;                        // t0: j=c
    else if (mode == 1) idx = b * CHW + (size_t)(l / NW) * (NH * NW) + j * NW + (l % NW); // t1: j=h
    else                idx = b * CHW + (size_t)(l / NH) * (NH * NW) + (l % NH) * NW + j; // t2: j=w

    const float v = x[idx];
    s1[j] = v; s2[j] = v * v;
    __syncthreads();
    for (int st = d >> 1; st > 0; st >>= 1) {
        if (j < st) { s1[j] += s1[j + st]; s2[j] += s2[j + st]; }
        __syncthreads();
    }
    const float mu  = s1[0] / d;
    const float var = s2[0] / d - mu * mu;
    out[(size_t)row * d + j] = (v - mu) * rsqrtf(var + 1e-5f) * lw[j] + lb[j];
}

// ---------------------------------------------------------------------
// Causal depthwise conv (k=4) along L over xi = xz[:, :din], then SiLU.
// ---------------------------------------------------------------------
__global__ void k_dwconv_silu(const float* __restrict__ xz,
                              const float* __restrict__ Wc,
                              const float* __restrict__ bc,
                              float* __restrict__ xc,
                              int R, int L, int din)
{
    const int idx = blockIdx.x * blockDim.x + threadIdx.x;
    if (idx >= R * din) return;
    const int d   = idx % din;
    const int row = idx / din;
    const int l   = row % L;
    const int base = row - l;
    float s = bc[d];
#pragma unroll
    for (int j = 0; j < 4; ++j) {
        const int ll = l - 3 + j;
        if (ll >= 0)
            s += Wc[d * 4 + j] * xz[(size_t)(base + ll) * (2 * din) + d];
    }
    xc[(size_t)row * din + d] = s / (1.f + __expf(-s));
}

// ---------------------------------------------------------------------
// Selective scan, fused dt projection + softplus + C-reduction + D skip
// + SiLU(z) gate. One thread per (b, d, n); shfl-reduce over the 16-state
// group. Prefetch (global_prefetch_b8) hides the xdbl stream latency.
// ---------------------------------------------------------------------
__global__ void k_scan(const float* __restrict__ xz,
                       const float* __restrict__ xc,
                       const float* __restrict__ xdbl,
                       const float* __restrict__ Wdt,
                       const float* __restrict__ bdt,
                       const float* __restrict__ A_log,
                       const float* __restrict__ Dp,
                       float* __restrict__ y,
                       int L, int din, int dtr)
{
    const int tid   = blockIdx.x * blockDim.x + threadIdx.x;
    const int chain = tid / NSTATE;
    const int n     = tid % NSTATE;
    if (chain >= NB * din) return;
    const int b = chain / din;
    const int d = chain % din;
    const int xw = dtr + 2 * NSTATE;

    const float Ac   = -__expf(A_log[d * NSTATE + n]);
    const float bdtd = bdt[d];
    const float Dpd  = Dp[d];
    float wdt[8];
#pragma unroll
    for (int r = 0; r < 8; ++r) wdt[r] = (r < dtr) ? Wdt[d * dtr + r] : 0.f;

    float h = 0.f;
    const size_t base = (size_t)b * L;
    for (int l = 0; l < L; ++l) {
        const size_t row = base + l;
        const float* xr = xdbl + row * xw;
        if (l + 8 < L) __builtin_prefetch(xdbl + (row + 8) * xw, 0, 1);

        float dtv = bdtd;
        for (int r = 0; r < dtr; ++r) dtv += xr[r] * wdt[r];
        dtv = (dtv > 20.f) ? dtv : log1pf(__expf(dtv));

        const float xcv = xc[row * din + d];
        const float Bv  = xr[dtr + n];
        const float Cv  = xr[dtr + NSTATE + n];
        h = __expf(dtv * Ac) * h + dtv * Bv * xcv;

        float p = h * Cv;                      // reduce over the 16 states
        p += __shfl_xor(p, 1);
        p += __shfl_xor(p, 2);
        p += __shfl_xor(p, 4);
        p += __shfl_xor(p, 8);
        if (n == 0) {
            const float zv = xz[row * (2 * din) + din + d];
            y[row * din + d] = (p + Dpd * xcv) * (zv / (1.f + __expf(-zv)));
        }
    }
}

// ---------------------------------------------------------------------
// attn = (y0+y1+y2)/3 mapped back to (B,C,H,W); yres = x + attn.
// ---------------------------------------------------------------------
__global__ void k_combine(const float* __restrict__ x,
                          const float* __restrict__ o0,
                          const float* __restrict__ o1,
                          const float* __restrict__ o2,
                          float* __restrict__ yres)
{
    const int idx = blockIdx.x * blockDim.x + threadIdx.x;
    if (idx >= NB * NC * NH * NW) return;
    const int b   = idx / (NC * NH * NW);
    const int rem = idx % (NC * NH * NW);
    const int c   = rem / (NH * NW);
    const int hw  = rem % (NH * NW);
    const int hh  = hw / NW, ww = hw % NW;

    const float a0 = o0[((size_t)b * NH * NW + hw) * NC + c];
    const float a1 = o1[((size_t)b * NC * NW + c * NW + ww) * NH + hh];
    const float a2 = o2[((size_t)b * NC * NH + c * NH + hh) * NW + ww];
    yres[idx] = x[idx] + (a0 + a1 + a2) * (1.f / 3.f);
}

// ---------------------------------------------------------------------
// LayerNorm over channels at each (b,h,w); write (pos, C) row-major.
// ---------------------------------------------------------------------
__global__ void k_ln_pos(const float* __restrict__ yres,
                         const float* __restrict__ lw,
                         const float* __restrict__ lb,
                         float* __restrict__ yn)
{
    __shared__ float s1[NC];
    __shared__ float s2[NC];
    const int pos = blockIdx.x;              // b*NH*NW + hw
    const int b   = pos / (NH * NW);
    const int hw  = pos % (NH * NW);
    const int c   = threadIdx.x;

    const float v = yres[(size_t)b * NC * NH * NW + (size_t)c * NH * NW + hw];
    s1[c] = v; s2[c] = v * v;
    __syncthreads();
    for (int st = NC >> 1; st > 0; st >>= 1) {
        if (c < st) { s1[c] += s1[c + st]; s2[c] += s2[c + st]; }
        __syncthreads();
    }
    const float mu  = s1[0] / NC;
    const float var = s2[0] / NC - mu * mu;
    yn[(size_t)pos * NC + c] = (v - mu) * rsqrtf(var + 1e-5f) * lw[c] + lb[c];
}

// ---------------------------------------------------------------------
// GDFN: depthwise 3x3 SAME on 340 channels (+bias), gelu(x1)*x2 gate.
// ---------------------------------------------------------------------
__global__ void k_dw3x3_gate(const float* __restrict__ hbuf,   // (pos,340)
                             const float* __restrict__ Wdw,    // (340,1,3,3)
                             const float* __restrict__ bdw,    // (340)
                             float* __restrict__ g)            // (pos,170)
{
    const int idx = blockIdx.x * blockDim.x + threadIdx.x;
    const int hid = 170;
    if (idx >= NB * NH * NW * hid) return;
    const int ch  = idx % hid;
    const int pos = idx / hid;
    const int b   = pos / (NH * NW);
    const int hw  = pos % (NH * NW);
    const int hh  = hw / NW, ww = hw % NW;

    float s1 = bdw[ch], s2 = bdw[ch + hid];
#pragma unroll
    for (int dy = 0; dy < 3; ++dy) {
#pragma unroll
        for (int dx = 0; dx < 3; ++dx) {
            const int yy = hh + dy - 1, xx = ww + dx - 1;
            if (yy >= 0 && yy < NH && xx >= 0 && xx < NW) {
                const size_t p2 = ((size_t)b * NH * NW + yy * NW + xx) * (2 * hid);
                s1 += Wdw[(size_t)ch * 9 + dy * 3 + dx]         * hbuf[p2 + ch];
                s2 += Wdw[(size_t)(ch + hid) * 9 + dy * 3 + dx] * hbuf[p2 + ch + hid];
            }
        }
    }
    const float ge = 0.5f * s1 * (1.f + erff(s1 * 0.70710678118654752f));
    g[(size_t)pos * hid + ch] = ge * s2;
}

__global__ void k_final(const float* __restrict__ yres,
                        const float* __restrict__ pw2,      // (pos,64)
                        float* __restrict__ out)
{
    const int idx = blockIdx.x * blockDim.x + threadIdx.x;
    if (idx >= NB * NC * NH * NW) return;
    const int b   = idx / (NC * NH * NW);
    const int rem = idx % (NC * NH * NW);
    const int c   = rem / (NH * NW);
    const int hw  = rem % (NH * NW);
    out[idx] = yres[idx] + pw2[((size_t)b * NH * NW + hw) * NC + c];
}

// =====================================================================
extern "C" void kernel_launch(void* const* d_in, const int* in_sizes, int n_in,
                              void* d_out, int out_size, void* d_ws, size_t ws_size,
                              hipStream_t stream)
{
    (void)in_sizes; (void)n_in; (void)out_size; (void)ws_size;
    const float* x = (const float*)d_in[0];
    // m1
    const float *Win1 = (const float*)d_in[1],  *Wc1  = (const float*)d_in[2];
    const float *bc1  = (const float*)d_in[3],  *Wx1  = (const float*)d_in[4];
    const float *Wdt1 = (const float*)d_in[5],  *bdt1 = (const float*)d_in[6];
    const float *Al1  = (const float*)d_in[7],  *Dp1  = (const float*)d_in[8];
    const float *Wo1  = (const float*)d_in[9];
    // m2
    const float *Win2 = (const float*)d_in[10], *Wc2  = (const float*)d_in[11];
    const float *bc2  = (const float*)d_in[12], *Wx2  = (const float*)d_in[13];
    const float *Wdt2 = (const float*)d_in[14], *bdt2 = (const float*)d_in[15];
    const float *Al2  = (const float*)d_in[16], *Dp2  = (const float*)d_in[17];
    const float *Wo2  = (const float*)d_in[18];
    // norms / ffn
    const float *ln1w = (const float*)d_in[19], *ln1b = (const float*)d_in[20];
    const float *ln2w = (const float*)d_in[21], *ln2b = (const float*)d_in[22];
    const float *lnfw = (const float*)d_in[23], *lnfb = (const float*)d_in[24];
    const float *Wpi  = (const float*)d_in[25], *Wdw  = (const float*)d_in[26];
    const float *bdw  = (const float*)d_in[27], *Wpo  = (const float*)d_in[28];

    // ---------------- workspace layout (floats) ----------------
    float* wsf   = (float*)d_ws;
    float* normA = wsf;                      // 2,097,152
    float* xz    = normA + 2097152;          // 8,388,608
    float* xc    = xz    + 8388608;          // 4,194,304
    float* xdbl  = xc    + 4194304;          // 1,310,720
    float* yseq  = xdbl  + 1310720;          // 4,194,304  (region ends 20,185,088)
    // GDFN overlay of the same region (mamba scratch is dead by then):
    float* hbuf  = wsf;                      // 11,141,120 (32768 x 340)
    float* gbuf  = hbuf + 11141120;          //  5,570,560 (32768 x 170)
    float* pw2   = gbuf + 5570560;           //  2,097,152
    // Persistent:
    float* out0  = wsf  + 20185088;          // 2,097,152
    float* out1  = out0 + 2097152;
    float* out2  = out1 + 2097152;
    float* yres  = out2 + 2097152;
    float* yn    = yres + 2097152;           // total ~30.7M floats (~123 MB)

    auto run_branch = [&](int mode, int L, int dmodel, int din, int dtr,
                          const float* lw, const float* lb,
                          const float* Win, const float* Wcv, const float* bcv,
                          const float* Wx,  const float* Wdt, const float* bdt,
                          const float* Al,  const float* Dp,  const float* Wout,
                          float* outbuf)
    {
        const int R = NB * L;
        k_gather_ln<<<R, dmodel, 0, stream>>>(x, lw, lb, normA, L, dmodel, mode);

        dim3 g1((R + 15) / 16, (2 * din + 31) / 32);
        k_gemm_wmma<<<g1, 32, 0, stream>>>(normA, Win, nullptr, xz, R, 2 * din, dmodel);

        const int ct = R * din;
        k_dwconv_silu<<<(ct + 255) / 256, 256, 0, stream>>>(xz, Wcv, bcv, xc, R, L, din);

        const int xw = dtr + 2 * NSTATE;
        dim3 g2((R + 15) / 16, (xw + 31) / 32);
        k_gemm_wmma<<<g2, 32, 0, stream>>>(xc, Wx, nullptr, xdbl, R, xw, din);

        const int chains = NB * din * NSTATE;
        k_scan<<<(chains + 255) / 256, 256, 0, stream>>>(xz, xc, xdbl, Wdt, bdt,
                                                         Al, Dp, yseq, L, din, dtr);

        dim3 g3((R + 15) / 16, (dmodel + 31) / 32);
        k_gemm_wmma<<<g3, 32, 0, stream>>>(yseq, Wout, nullptr, outbuf, R, dmodel, din);
    };

    // Branch 0: sequence over H*W, d=64, m1
    run_branch(0, NH * NW, 64, 128, 4, ln1w, ln1b,
               Win1, Wc1, bc1, Wx1, Wdt1, bdt1, Al1, Dp1, Wo1, out0);
    // Branch 1: sequence over C*W, features = H (d=128), m2
    run_branch(1, NC * NW, 128, 256, 8, ln2w, ln2b,
               Win2, Wc2, bc2, Wx2, Wdt2, bdt2, Al2, Dp2, Wo2, out1);
    // Branch 2: sequence over C*H, features = W (d=128), m2
    run_branch(2, NC * NH, 128, 256, 8, ln2w, ln2b,
               Win2, Wc2, bc2, Wx2, Wdt2, bdt2, Al2, Dp2, Wo2, out2);

    // Combine + residual, LN, GDFN
    const int total = NB * NC * NH * NW;
    k_combine<<<(total + 255) / 256, 256, 0, stream>>>(x, out0, out1, out2, yres);

    const int npos = NB * NH * NW;
    k_ln_pos<<<npos, NC, 0, stream>>>(yres, lnfw, lnfb, yn);

    dim3 gp1((npos + 15) / 16, (340 + 31) / 32);
    k_gemm_wmma<<<gp1, 32, 0, stream>>>(yn, Wpi, nullptr, hbuf, npos, 340, NC);

    const int gt = npos * 170;
    k_dw3x3_gate<<<(gt + 255) / 256, 256, 0, stream>>>(hbuf, Wdw, bdw, gbuf);

    dim3 gp2((npos + 15) / 16, (NC + 31) / 32);
    k_gemm_wmma<<<gp2, 32, 0, stream>>>(gbuf, Wpo, nullptr, pw2, npos, NC, 170);

    k_final<<<(total + 255) / 256, 256, 0, stream>>>(yres, pw2, (float*)d_out);
}
